// Model_39676907881404
// MI455X (gfx1250) — matrix operand order repaired and checked by reading, not compiled, not run on hardware
//
#include <hip/hip_runtime.h>

typedef __attribute__((ext_vector_type(16))) __bf16 v16bf;
typedef __attribute__((ext_vector_type(8)))  float  v8f;
typedef __attribute__((ext_vector_type(4)))  float  f4;

namespace {
constexpr int S     = 2048;
constexpr int D     = 128;
constexpr int QTILE = 128;   // q rows per block (8 waves x 16)
constexpr int KTILE = 32;    // keys per main-loop iteration
constexpr int NT    = S / KTILE;
constexpr int KSTR  = 136;   // LDS row stride, K tile (bf16 elems), 16B-mult
constexpr int VSTR  = 40;    // LDS row stride, V^T tile (bf16 elems), 16B-mult
constexpr int PSTR  = 40;    // LDS row stride, P scratch (bf16 elems), 16B-mult

union Pack16 { uint4 u[2]; v16bf v; };
union Pack8  { uint4 u;    __bf16 h[8]; };

// 1-multiply xorshift: deterministic dropout mask source
__device__ __forceinline__ unsigned rng32(unsigned x) {
  x ^= x >> 15; x *= 2654435761u; x ^= x >> 13; return x;
}
} // namespace

__global__ __launch_bounds__(256) void attn_fwd_bf16wmma(
    const float* __restrict__ Q, const float* __restrict__ Kp,
    const float* __restrict__ Vp, const float* __restrict__ scale_p,
    const float* __restrict__ drop_p, float* __restrict__ Out)
{
  __shared__ __bf16 ldsK[KTILE * KSTR];   // K tile, row-major bf16
  __shared__ __bf16 ldsV[D * VSTR];       // V tile, transposed bf16 (Vt[d][k])
  __shared__ __bf16 ldsP[8 * 16 * PSTR];  // per-wave P scratch, bf16 [q][k]

  const int tid  = threadIdx.x;
  const int lane = tid & 31;
  const int wave = tid >> 5;
  const int lh   = lane & 15;
  const int hi   = lane >> 4;

  const int bh    = blockIdx.x >> 4;   // 0..31 (B*H)
  const int qb    = blockIdx.x & 15;   // 0..15
  const int qbase = qb * QTILE + wave * 16;

  const float scale    = scale_p[0];
  const float pdrop    = drop_p[0];
  const float inv_keep = 1.0f / (1.0f - pdrop);
  const unsigned keep_thresh = (unsigned)((1.0 - (double)pdrop) * 4294967295.0);

  const size_t base = (size_t)bh * S * D;

  // ---- preload this wave's 16 Q rows into bf16 WMMA A-layout, pre-scaled ----
  // A layout (16-bit, 16x32): lane<16 row M=lh holds K {0..7,16..23};
  // lane>=16 holds K {8..15,24..31}; VGPR j packs pair (2j,2j+1).
  v16bf qa[4];
  {
    const float* qrow = Q + base + (size_t)(qbase + lh) * D;
    #pragma unroll
    for (int s = 0; s < 4; ++s) {
      const int d0 = 32 * s + hi * 8;
      f4 f0 = *(const f4*)(qrow + d0);
      f4 f1 = *(const f4*)(qrow + d0 + 4);
      f4 f2 = *(const f4*)(qrow + d0 + 16);
      f4 f3 = *(const f4*)(qrow + d0 + 20);
      #pragma unroll
      for (int e = 0; e < 4; ++e) {
        qa[s][e]      = (__bf16)(f0[e] * scale);
        qa[s][4 + e]  = (__bf16)(f1[e] * scale);
        qa[s][8 + e]  = (__bf16)(f2[e] * scale);
        qa[s][12 + e] = (__bf16)(f3[e] * scale);
      }
    }
  }

  // O accumulators: 8 d-tiles of 16x16 f32 (elem j <-> row j+8*hi, col lh)
  v8f o[8];
  #pragma unroll
  for (int t = 0; t < 8; ++t) {
    #pragma unroll
    for (int j = 0; j < 8; ++j) o[t][j] = 0.0f;
  }
  float m[8], l[8];
  #pragma unroll
  for (int j = 0; j < 8; ++j) { m[j] = -1e30f; l[j] = 0.0f; }

  __bf16* pScr = &ldsP[wave * 16 * PSTR];

  const int sr = tid >> 3;         // staging row within K/V tile: 0..31
  const int sc = (tid & 7) * 16;   // staging col: 0,16,...,112
  const float* kbase = Kp + base + (size_t)sr * D + sc;
  const float* vbase = Vp + base + (size_t)sr * D + sc;

  // ---- software pipeline: prefetch tile 0 into registers
  f4 kreg[4], vreg[4];
  #pragma unroll
  for (int e = 0; e < 4; ++e) {
    kreg[e] = *(const f4*)(kbase + 4 * e);
    vreg[e] = *(const f4*)(vbase + 4 * e);
  }

  for (int kt = 0; kt < NT; ++kt) {
    __syncthreads();   // all waves done reading previous K/V LDS tiles

    // ---- convert prefetched registers -> bf16 LDS (K row-major, V transposed)
    {
      Pack8 p0, p1;
      #pragma unroll
      for (int e = 0; e < 4; ++e) {
        p0.h[e] = (__bf16)kreg[0][e]; p0.h[4 + e] = (__bf16)kreg[1][e];
        p1.h[e] = (__bf16)kreg[2][e]; p1.h[4 + e] = (__bf16)kreg[3][e];
      }
      *(uint4*)&ldsK[sr * KSTR + sc]     = p0.u;
      *(uint4*)&ldsK[sr * KSTR + sc + 8] = p1.u;
      #pragma unroll
      for (int e = 0; e < 4; ++e) {
        ldsV[(sc + e)      * VSTR + sr] = (__bf16)vreg[0][e];
        ldsV[(sc + 4 + e)  * VSTR + sr] = (__bf16)vreg[1][e];
        ldsV[(sc + 8 + e)  * VSTR + sr] = (__bf16)vreg[2][e];
        ldsV[(sc + 12 + e) * VSTR + sr] = (__bf16)vreg[3][e];
      }
    }
    __syncthreads();

    // ---- issue global loads for tile kt+1 now; they complete during compute
    if (kt + 1 < NT) {
      const float* kn = kbase + (size_t)(kt + 1) * KTILE * D;
      const float* vn = vbase + (size_t)(kt + 1) * KTILE * D;
      #pragma unroll
      for (int e = 0; e < 4; ++e) {
        kreg[e] = *(const f4*)(kn + 4 * e);
        vreg[e] = *(const f4*)(vn + 4 * e);
      }
      if (kt + 2 < NT) {   // L2 prefetch two tiles ahead
        __builtin_prefetch(kn + KTILE * D, 0, 1);
        __builtin_prefetch(vn + KTILE * D, 0, 1);
      }
    }

    // ---- scores: 16 q-rows x 32 keys
    v8f c0, c1;
    #pragma unroll
    for (int j = 0; j < 8; ++j) { c0[j] = 0.0f; c1[j] = 0.0f; }
    #pragma unroll
    for (int s = 0; s < 4; ++s) {
      const __bf16* kp0 = &ldsK[lh * KSTR + 32 * s + 8 * hi];          // keys 0..15
      const __bf16* kp1 = &ldsK[(16 + lh) * KSTR + 32 * s + 8 * hi];   // keys 16..31
      Pack16 kb0, kb1;
      kb0.u[0] = *(const uint4*)(kp0); kb0.u[1] = *(const uint4*)(kp0 + 16);
      kb1.u[0] = *(const uint4*)(kp1); kb1.u[1] = *(const uint4*)(kp1 + 16);
      c0 = __builtin_amdgcn_wmma_f32_16x16x32_bf16(false, qa[s], false, kb0.v,
                                                   (short)0, c0, false, false);
      c1 = __builtin_amdgcn_wmma_f32_16x16x32_bf16(false, qa[s], false, kb1.v,
                                                   (short)0, c1, false, false);
    }

    // ---- online softmax + dropout; write unnormalized dropped P (bf16) to LDS
    const int k0 = kt * KTILE;
    const unsigned qglob_base = (unsigned)bh * (unsigned)S;
    #pragma unroll
    for (int j = 0; j < 8; ++j) {
      float tmax = fmaxf(c0[j], c1[j]);
      #pragma unroll
      for (int sh = 1; sh < 16; sh <<= 1) tmax = fmaxf(tmax, __shfl_xor(tmax, sh, 32));
      const float mn    = fmaxf(m[j], tmax);
      const float alpha = __expf(m[j] - mn);
      const float p0    = __expf(c0[j] - mn);
      const float p1    = __expf(c1[j] - mn);
      float rs = p0 + p1;
      #pragma unroll
      for (int sh = 1; sh < 16; sh <<= 1) rs += __shfl_xor(rs, sh, 32);
      l[j] = l[j] * alpha + rs;
      m[j] = mn;
      #pragma unroll
      for (int t = 0; t < 8; ++t) o[t][j] *= alpha;

      const int row = j + 8 * hi;
      const unsigned q    = (unsigned)(qbase + row);
      const unsigned idx0 = (qglob_base + q) * (unsigned)S + (unsigned)(k0 + lh);
      const float pd0 = (rng32(idx0)       < keep_thresh) ? p0 * inv_keep : 0.0f;
      const float pd1 = (rng32(idx0 + 16u) < keep_thresh) ? p1 * inv_keep : 0.0f;
      pScr[row * PSTR + lh]      = (__bf16)pd0;
      pScr[row * PSTR + 16 + lh] = (__bf16)pd1;
    }

    // ---- reload P in A-layout (wave-private LDS; 2x ds_load_b128, no reconvert)
    v16bf pa;
    {
      const __bf16* pr = pScr + lh * PSTR + hi * 8;
      Pack16 pp;
      pp.u[0] = *(const uint4*)(pr);        // K = hi*8 + 0..7
      pp.u[1] = *(const uint4*)(pr + 16);   // K = 16 + hi*8 + 0..7
      pa = pp.v;
    }

    // ---- O += P * V  (B from transposed V in LDS: K = hi*8+0..7, 16+hi*8+0..7)
    #pragma unroll
    for (int t = 0; t < 8; ++t) {
      const __bf16* vp = &ldsV[(16 * t + lh) * VSTR + 8 * hi];
      Pack16 vb;
      vb.u[0] = *(const uint4*)(vp);
      vb.u[1] = *(const uint4*)(vp + 16);
      o[t] = __builtin_amdgcn_wmma_f32_16x16x32_bf16(false, pa, false, vb.v,
                                                     (short)0, o[t], false, false);
    }
  }

  // ---- epilogue: normalize by softmax denominator and store fp32
  float* obase = Out + base + (size_t)qbase * D;
  #pragma unroll
  for (int j = 0; j < 8; ++j) {
    const int row    = j + 8 * hi;
    const float invl = 1.0f / l[j];
    #pragma unroll
    for (int t = 0; t < 8; ++t)
      obase[(size_t)row * D + 16 * t + lh] = o[t][j] * invl;
  }
}

extern "C" void kernel_launch(void* const* d_in, const int* in_sizes, int n_in,
                              void* d_out, int out_size, void* d_ws, size_t ws_size,
                              hipStream_t stream) {
  (void)in_sizes; (void)n_in; (void)out_size; (void)d_ws; (void)ws_size;
  const float* q  = (const float*)d_in[0];
  const float* k  = (const float*)d_in[1];
  const float* v  = (const float*)d_in[2];
  const float* sc = (const float*)d_in[3];
  const float* dp = (const float*)d_in[4];
  float* out = (float*)d_out;
  // grid = (B*H) * (S / QTILE) = 32 * 16 = 512 blocks, 256 threads (8 wave32)
  attn_fwd_bf16wmma<<<dim3(512), dim3(256), 0, stream>>>(q, k, v, sc, dp, out);
}
